// MPNNPOM_32839319945357
// MI455X (gfx1250) — matrix operand order repaired and checked by reading, not compiled, not run on hardware
//
#include <hip/hip_runtime.h>
#include <hip/hip_bf16.h>
#include <cstdint>
#include <cstddef>

// ---------------------------------------------------------------------------
// MPNN-POM forward for MI455X (gfx1250): bf16 WMMA GEMMs + atomic scatter.
// GEMM inner loop is branch-free: A/B fragments are 128-bit vector loads from
// row-major A [M,K] and pre-transposed B [Npad,Kpad] (weights converted once).
// ---------------------------------------------------------------------------

typedef __attribute__((ext_vector_type(16))) __bf16 v16bf;
typedef __attribute__((ext_vector_type(8)))  float  v8f;

static __device__ __forceinline__ unsigned short f2bf(float f) {
  unsigned int u = __float_as_uint(f);
  u += 0x7fffu + ((u >> 16) & 1u);               // round-to-nearest-even
  return (unsigned short)(u >> 16);
}
static __device__ __forceinline__ float bf2f(unsigned short h) {
  return __uint_as_float(((unsigned int)h) << 16);
}
static __device__ __forceinline__ float sigmoidf_(float x) {
  return 1.0f / (1.0f + __expf(-x));
}
// monotonic float -> uint key for atomicMax-based segment max
static __device__ __forceinline__ unsigned int fkey(float f) {
  unsigned int u = __float_as_uint(f);
  return (u & 0x80000000u) ? ~u : (u | 0x80000000u);
}
static __device__ __forceinline__ float funkey(unsigned int k) {
  return (k & 0x80000000u) ? __uint_as_float(k ^ 0x80000000u)
                           : __uint_as_float(~k);
}

// ---------------------------------------------------------------------------
// WMMA GEMM: C[M,N] = A[M,K] @ Bt[N,K]^T (+bias)(+=C)(relu)
//  - one 16x16 tile per wave32, K stepped by 32 (v_wmma_f32_16x16x32_bf16)
//  - REQUIRES: M % 16 == 0, Npad % 16 == 0, K % 32 == 0 (buffers padded)
//  - A row-major bf16 [M,K]; Bt transposed bf16 [Npad,K] (zero-padded)
//  - epilogue: cols >= Nreal written as 0 (if < Nstore); ldc = C row stride
// Fragment layouts per CDNA5 ISA 7.12.2 (wave32):
//  A: lane(0-15)=row M, lanes 0-15 carry K 0..7/16..23, lanes 16-31 K 8..15/24..31
//  B: lane=col N, lanes 0-15 carry K 0..7/16..23, lanes 16-31 K 8..15/24..31 (as A)
//  C/D: VGPR r -> M = r + 8*(lane>=16), N = lane & 15
// ---------------------------------------------------------------------------
__global__ void k_wmma_gemm(const unsigned short* __restrict__ A,
                            const unsigned short* __restrict__ Bt,
                            const float* __restrict__ bias,
                            float* __restrict__ Cf,
                            unsigned short* __restrict__ Cb,
                            int M, int K, int tilesN,
                            int Nreal, int Nstore, int ldc,
                            int relu, int accum)
{
  int wave = blockIdx.x * (blockDim.x >> 5) + (threadIdx.x >> 5);
  int lane = threadIdx.x & 31;
  int tM = wave / tilesN;
  int tN = wave - tM * tilesN;
  if (tM * 16 >= M) return;                      // wave-uniform exit

  int mloc = lane & 15;
  int ksel = lane >> 4;

  const unsigned short* Arow = A  + (size_t)(tM * 16 + mloc) * K + ksel * 8;
  const unsigned short* Brow = Bt + (size_t)(tN * 16 + mloc) * K + ksel * 8;

  v8f acc = {};
  for (int k0 = 0; k0 < K; k0 += 32) {
    union { uint4 q[2]; v16bf v; } fa, fb;
    fa.q[0] = *(const uint4*)(Arow);             // K = base .. base+7
    fa.q[1] = *(const uint4*)(Arow + 16);        // K = base+16 .. base+23
    fb.q[0] = *(const uint4*)(Brow);             // K = base .. base+7
    fb.q[1] = *(const uint4*)(Brow + 16);        // K = base+16 .. base+23
    acc = __builtin_amdgcn_wmma_f32_16x16x32_bf16(false, fa.v, false, fb.v,
                                                  (short)0, acc, false, false);
    Arow += 32; Brow += 32;
  }

#pragma unroll
  for (int r = 0; r < 8; ++r) {
    int m = tM * 16 + r + ksel * 8;              // M % 16 == 0 -> always in range
    int n = tN * 16 + (lane & 15);
    if (n < Nstore) {
      size_t idx = (size_t)m * ldc + n;
      float v = 0.0f;
      if (n < Nreal) {
        v = acc[r];
        if (bias)  v += bias[n];
        if (accum && Cf) v += Cf[idx];
        if (relu)  v = fmaxf(v, 0.0f);
      }
      if (Cf) Cf[idx] = v;
      if (Cb) Cb[idx] = f2bf(v);
    }
  }
}

// ---------------------------------------------------------------------------
// Weight convert+transpose: src f32 [Kreal, Nreal] -> dst bf16 [Npad, Kpad]
// dst[n][k] = src[k][col(n)], zero-padded. swap64: col = (n&63)<<6 | n>>6
// (permutes We2 so the edge GEMM emits per-edge weights in [o,i] layout).
// ---------------------------------------------------------------------------
__global__ void k_f2bt(const float* __restrict__ src, unsigned short* __restrict__ dst,
                       int Nreal, int Kreal, int Npad, int Kpad, int swap64)
{
  int t = blockIdx.x * blockDim.x + threadIdx.x;
  if (t >= Npad * Kpad) return;
  int n = t / Kpad, k = t - n * Kpad;
  unsigned short v = 0;
  if (n < Nreal && k < Kreal) {
    int col = swap64 ? (((n & 63) << 6) | (n >> 6)) : n;
    v = f2bf(src[(size_t)k * Nreal + col]);
  }
  dst[t] = v;
}

// ---------------------------------------------------------------------------
// Small elementwise / scatter kernels
// ---------------------------------------------------------------------------
__global__ void k_zero32(unsigned int* p, size_t n) {
  size_t t = (size_t)blockIdx.x * blockDim.x + threadIdx.x;
  if (t < n) p[t] = 0u;
}
__global__ void k_f2b(const float* __restrict__ s, unsigned short* __restrict__ d, size_t n) {
  size_t t = (size_t)blockIdx.x * blockDim.x + threadIdx.x;
  if (t < n) d[t] = f2bf(s[t]);
}
__global__ void k_init_negmax(unsigned int* p, int n) {
  int t = blockIdx.x * blockDim.x + threadIdx.x;
  if (t < n) p[t] = fkey(-3.402823466e38f);
}

// h = relu(node_feats @ Wn + bn), write f32 + bf16
__global__ void k_node_proj(const float* __restrict__ nf, const float* __restrict__ Wn,
                            const float* __restrict__ bn, float* __restrict__ h,
                            unsigned short* __restrict__ hb, int Nn)
{
  int t = blockIdx.x * blockDim.x + threadIdx.x;
  if (t >= Nn * 64) return;
  int n = t >> 6, o = t & 63;
  const float* x = nf + (size_t)n * 134;
  float acc = bn[o];
  for (int j = 0; j < 134; ++j) acc += x[j] * Wn[j * 64 + o];
  acc = fmaxf(acc, 0.0f);
  h[t] = acc; hb[t] = f2bf(acc);
}

// hidden = relu(edge_feats @ We1 + be1)  -> bf16 [E,128]
__global__ void k_edge_hidden(const float* __restrict__ ef, const float* __restrict__ W,
                              const float* __restrict__ b, unsigned short* __restrict__ out,
                              int E)
{
  int t = blockIdx.x * blockDim.x + threadIdx.x;
  if (t >= E * 128) return;
  int e = t >> 7, k = t & 127;
  const float* x = ef + (size_t)e * 6;
  float acc = b[k];
  for (int j = 0; j < 6; ++j) acc += x[j] * W[j * 128 + k];
  out[t] = f2bf(fmaxf(acc, 0.0f));
}

// e_emb = relu(edge_feats @ Wep + bep)  -> f32 [E,64]
__global__ void k_edge_emb(const float* __restrict__ ef, const float* __restrict__ W,
                           const float* __restrict__ b, float* __restrict__ out, int E)
{
  int t = blockIdx.x * blockDim.x + threadIdx.x;
  if (t >= E * 64) return;
  int e = t >> 6, k = t & 63;
  const float* x = ef + (size_t)e * 6;
  float acc = b[k];
  for (int j = 0; j < 6; ++j) acc += x[j] * W[j * 64 + k];
  out[t] = fmaxf(acc, 0.0f);
}

// per-edge matvec: msg[e,o] = sum_i h[src[e],i] * Wt[e,o,i]; scatter-add to agg[dst]
// Wt stored [e][o][i] (o-major) so each lane streams 128B contiguous per output.
__global__ void k_msg_scatter(const unsigned short* __restrict__ Wt,
                              const float* __restrict__ h,
                              const int* __restrict__ src, const int* __restrict__ dst,
                              float* __restrict__ agg, int E)
{
  int wave = blockIdx.x * (blockDim.x >> 5) + (threadIdx.x >> 5);
  if (wave >= E) return;
  int lane = threadIdx.x & 31;
  int s = src[wave], d = dst[wave];
  const float* hr = h + (size_t)s * 64;
  const unsigned short* w0 = Wt + (size_t)wave * 4096 + lane * 64;        // o = lane
  const unsigned short* w1 = w0 + 32 * 64;                                // o = lane+32
  if (wave + 1 < E) __builtin_prefetch(Wt + (size_t)(wave + 1) * 4096, 0, 0);
  float a0 = 0.0f, a1 = 0.0f;
#pragma unroll
  for (int c = 0; c < 8; ++c) {
    uint4 q0 = *(const uint4*)(w0 + c * 8);
    uint4 q1 = *(const uint4*)(w1 + c * 8);
    const unsigned int* u0 = (const unsigned int*)&q0;
    const unsigned int* u1 = (const unsigned int*)&q1;
#pragma unroll
    for (int j = 0; j < 4; ++j) {
      float h0 = hr[c * 8 + 2 * j], h1 = hr[c * 8 + 2 * j + 1];
      a0 += h0 * __uint_as_float(u0[j] << 16) + h1 * __uint_as_float(u0[j] & 0xffff0000u);
      a1 += h0 * __uint_as_float(u1[j] << 16) + h1 * __uint_as_float(u1[j] & 0xffff0000u);
    }
  }
  atomicAdd(&agg[(size_t)d * 64 + lane], a0);
  atomicAdd(&agg[(size_t)d * 64 + lane + 32], a1);
}

// m = relu(agg + hres) -> bf16
__global__ void k_gru_m(const float* __restrict__ agg, const float* __restrict__ hres,
                        unsigned short* __restrict__ mb, size_t cnt)
{
  size_t t = (size_t)blockIdx.x * blockDim.x + threadIdx.x;
  if (t < cnt) mb[t] = f2bf(fmaxf(agg[t] + hres[t], 0.0f));
}

// GRU cell update; h (=hs) written f32 + bf16
__global__ void k_gru_update(const float* __restrict__ gx, const float* __restrict__ gh,
                             float* __restrict__ h, unsigned short* __restrict__ hb, int Nn)
{
  int t = blockIdx.x * blockDim.x + threadIdx.x;
  if (t >= Nn * 64) return;
  int n = t >> 6, c = t & 63;
  size_t b = (size_t)n * 192;
  float xr = gx[b + c], xz = gx[b + 64 + c], xn = gx[b + 128 + c];
  float hr = gh[b + c], hz = gh[b + 64 + c], hn = gh[b + 128 + c];
  float r = sigmoidf_(xr + hr);
  float z = sigmoidf_(xz + hz);
  float nn = tanhf(xn + r * hn);
  float hs = (1.0f - z) * nn + z * h[t];
  h[t] = hs; hb[t] = f2bf(hs);
}

// feat[dst] += concat(h[src], e_emb[e])
__global__ void k_readout(const float* __restrict__ h, const float* __restrict__ eemb,
                          const int* __restrict__ src, const int* __restrict__ dst,
                          float* __restrict__ feat, int E)
{
  int t = blockIdx.x * blockDim.x + threadIdx.x;
  if (t >= E * 128) return;
  int e = t >> 7, c = t & 127;
  float v = (c < 64) ? h[(size_t)src[e] * 64 + c] : eemb[(size_t)e * 64 + (c - 64)];
  atomicAdd(&feat[(size_t)dst[e] * 128 + c], v);
}

// LSTM cell (gates order i,f,g,o); hl written f32 + bf16
__global__ void k_lstm_cell(const float* __restrict__ g, float* __restrict__ hl,
                            float* __restrict__ cl, unsigned short* __restrict__ hlb, int Bb)
{
  int t = blockIdx.x * blockDim.x + threadIdx.x;
  if (t >= Bb * 128) return;
  int b = t >> 7, c = t & 127;
  size_t base = (size_t)b * 512;
  float gi = g[base + c], gf = g[base + 128 + c], gg = g[base + 256 + c], go = g[base + 384 + c];
  float cn = sigmoidf_(gf) * cl[t] + sigmoidf_(gi) * tanhf(gg);
  float hn = sigmoidf_(go) * tanhf(cn);
  cl[t] = cn; hl[t] = hn; hlb[t] = f2bf(hn);
}

// e[n] = feat[n].q[gid[n]] ; segment max via keyed atomicMax
__global__ void k_attn_dot(const float* __restrict__ feat, const float* __restrict__ q,
                           const int* __restrict__ gid, float* __restrict__ ev,
                           unsigned int* __restrict__ emax, int Nn)
{
  int n = blockIdx.x * blockDim.x + threadIdx.x;
  if (n >= Nn) return;
  int g = gid[n];
  const float* f = feat + (size_t)n * 128;
  const float* qq = q + (size_t)g * 128;
  float acc = 0.0f;
  for (int c = 0; c < 128; ++c) acc += f[c] * qq[c];
  ev[n] = acc;
  atomicMax(&emax[g], fkey(acc));
}

__global__ void k_attn_exp(const float* __restrict__ ev, const unsigned int* __restrict__ emax,
                           const int* __restrict__ gid, float* __restrict__ av,
                           float* __restrict__ denom, int Nn)
{
  int n = blockIdx.x * blockDim.x + threadIdx.x;
  if (n >= Nn) return;
  int g = gid[n];
  float a = __expf(ev[n] - funkey(emax[g]));
  av[n] = a;
  atomicAdd(&denom[g], a);
}

__global__ void k_attn_r(const float* __restrict__ feat, const float* __restrict__ av,
                         const float* __restrict__ denom, const int* __restrict__ gid,
                         float* __restrict__ r, int Nn)
{
  int t = blockIdx.x * blockDim.x + threadIdx.x;
  if (t >= Nn * 128) return;
  int n = t >> 7, c = t & 127;
  int g = gid[n];
  float w = av[n] / (denom[g] + 1e-9f);
  atomicAdd(&r[(size_t)g * 128 + c], w * feat[t]);
}

// q_star = concat(q, r) -> f32 + bf16
__global__ void k_qstar(const float* __restrict__ hl2, const float* __restrict__ r,
                        float* __restrict__ qs, unsigned short* __restrict__ qsb, int Bb)
{
  int t = blockIdx.x * blockDim.x + threadIdx.x;
  if (t >= Bb * 256) return;
  int b = t >> 8, c = t & 255;
  float v = (c < 128) ? hl2[(size_t)b * 128 + c] : r[(size_t)b * 128 + (c - 128)];
  qs[t] = v; qsb[t] = f2bf(v);
}

// comb = concat(q_star, relu(BN(fpproj)))  -> bf16 [B,384]
__global__ void k_build_comb(const float* __restrict__ qs, const float* __restrict__ fpproj,
                             const float* __restrict__ gma, const float* __restrict__ bta,
                             const float* __restrict__ mu, const float* __restrict__ var,
                             unsigned short* __restrict__ comb, int Bb)
{
  int t = blockIdx.x * blockDim.x + threadIdx.x;
  if (t >= Bb * 384) return;
  int b = t / 384, c = t - b * 384;
  float v;
  if (c < 256) {
    v = qs[(size_t)b * 256 + c];
  } else {
    int j = c - 256;
    float x = fpproj[(size_t)b * 128 + j];
    v = fmaxf((x - mu[j]) * gma[j] * rsqrtf(var[j] + 1e-5f) + bta[j], 0.0f);
  }
  comb[t] = f2bf(v);
}

// ---------------------------------------------------------------------------
// Host-side launch
// ---------------------------------------------------------------------------
static inline int nb(size_t n) { return (int)((n + 255) / 256); }

static void gemm(hipStream_t s, const unsigned short* A, const unsigned short* Bt,
                 const float* bias, float* Cf, unsigned short* Cb,
                 int M, int Npad, int K, int Nreal, int Nstore, int ldc,
                 int relu, int accum)
{
  int tilesM = M / 16, tilesN = Npad / 16;
  long waves = (long)tilesM * tilesN;
  int blocks = (int)((waves + 7) / 8);           // 8 waves per 256-thread block
  k_wmma_gemm<<<blocks, 256, 0, s>>>(A, Bt, bias, Cf, Cb, M, K, tilesN,
                                     Nreal, Nstore, ldc, relu, accum);
}

extern "C" void kernel_launch(void* const* d_in, const int* in_sizes, int n_in,
                              void* d_out, int out_size, void* d_ws, size_t ws_size,
                              hipStream_t stream)
{
  const float* node_feats = (const float*)d_in[0];
  const float* edge_feats = (const float*)d_in[1];
  const float* fp_vector  = (const float*)d_in[2];
  const int*   src        = (const int*)d_in[3];
  const int*   dst        = (const int*)d_in[4];
  const int*   gid        = (const int*)d_in[5];
  const float* Wn   = (const float*)d_in[6];   const float* bn_   = (const float*)d_in[7];
  const float* We1  = (const float*)d_in[8];   const float* be1   = (const float*)d_in[9];
  const float* We2  = (const float*)d_in[10];  const float* be2   = (const float*)d_in[11];
  const float* Wres = (const float*)d_in[12];
  const float* Wgx  = (const float*)d_in[13];  const float* bgx   = (const float*)d_in[14];
  const float* Wgh  = (const float*)d_in[15];  const float* bgh   = (const float*)d_in[16];
  const float* Wep  = (const float*)d_in[17];  const float* bep   = (const float*)d_in[18];
  const float* Wlx[3] = {(const float*)d_in[19], (const float*)d_in[22], (const float*)d_in[25]};
  const float* Wlh[3] = {(const float*)d_in[20], (const float*)d_in[23], (const float*)d_in[26]};
  const float* bl[3]  = {(const float*)d_in[21], (const float*)d_in[24], (const float*)d_in[27]};
  const float* Wfp  = (const float*)d_in[28];  const float* bfp   = (const float*)d_in[29];
  const float* bn_g = (const float*)d_in[30];  const float* bn_b  = (const float*)d_in[31];
  const float* bn_m = (const float*)d_in[32];  const float* bn_v  = (const float*)d_in[33];
  const float* Wf1  = (const float*)d_in[34];  const float* bf1   = (const float*)d_in[35];
  const float* Wf2  = (const float*)d_in[36];  const float* bf2   = (const float*)d_in[37];
  const float* Wp   = (const float*)d_in[38];  const float* bp    = (const float*)d_in[39];

  const int Nn = in_sizes[0] / 134;            // 20000
  const int E  = in_sizes[1] / 6;              // 60000
  const int Bb = in_sizes[2] / 1024;           // 800

  // ---- workspace carving (~600 MB) ----
  char* p = (char*)d_ws;
  auto carve = [&](size_t bytes) -> char* {
    char* r = p; p += (bytes + 255) & ~(size_t)255; return r;
  };
  // transposed bf16 weights [Npad, Kpad]
  unsigned short* We2b  = (unsigned short*)carve((size_t)4096 * 128 * 2);
  unsigned short* Wresb = (unsigned short*)carve((size_t)64 * 64 * 2);
  unsigned short* Wgxb  = (unsigned short*)carve((size_t)192 * 64 * 2);
  unsigned short* Wghb  = (unsigned short*)carve((size_t)192 * 64 * 2);
  unsigned short *Wlxb[3], *Wlhb[3];
  Wlxb[0] = (unsigned short*)carve((size_t)512 * 256 * 2);
  Wlhb[0] = (unsigned short*)carve((size_t)512 * 128 * 2);
  Wlxb[1] = (unsigned short*)carve((size_t)512 * 128 * 2);
  Wlhb[1] = (unsigned short*)carve((size_t)512 * 128 * 2);
  Wlxb[2] = (unsigned short*)carve((size_t)512 * 128 * 2);
  Wlhb[2] = (unsigned short*)carve((size_t)512 * 128 * 2);
  unsigned short* Wfpb  = (unsigned short*)carve((size_t)128 * 1024 * 2);
  unsigned short* Wf1b  = (unsigned short*)carve((size_t)320 * 384 * 2);
  unsigned short* Wf2b  = (unsigned short*)carve((size_t)256 * 320 * 2);
  unsigned short* Wpb   = (unsigned short*)carve((size_t)144 * 256 * 2);
  // activations
  unsigned short* fpb     = (unsigned short*)carve((size_t)Bb * 1024 * 2);
  unsigned short* hiddenb = (unsigned short*)carve((size_t)E * 128 * 2);
  float*          eemb    = (float*)carve((size_t)E * 64 * 4);
  unsigned short* Webf    = (unsigned short*)carve((size_t)E * 4096 * 2);  // 491.5 MB
  float*          h       = (float*)carve((size_t)Nn * 64 * 4);
  unsigned short* hb      = (unsigned short*)carve((size_t)Nn * 64 * 2);
  float*          agg     = (float*)carve((size_t)Nn * 64 * 4);
  float*          hres    = (float*)carve((size_t)Nn * 64 * 4);
  unsigned short* mb      = (unsigned short*)carve((size_t)Nn * 64 * 2);
  float*          gx      = (float*)carve((size_t)Nn * 192 * 4);
  float*          gh      = (float*)carve((size_t)Nn * 192 * 4);
  float*          feat    = (float*)carve((size_t)Nn * 128 * 4);
  float *hl[3], *cl[3]; unsigned short* hlb[3];
  for (int l = 0; l < 3; ++l) {
    hl[l]  = (float*)carve((size_t)Bb * 128 * 4);
    cl[l]  = (float*)carve((size_t)Bb * 128 * 4);
    hlb[l] = (unsigned short*)carve((size_t)Bb * 128 * 2);
  }
  float*          qs    = (float*)carve((size_t)Bb * 256 * 4);
  unsigned short* qsb   = (unsigned short*)carve((size_t)Bb * 256 * 2);
  float*          gbuf  = (float*)carve((size_t)Bb * 512 * 4);
  float*          ev    = (float*)carve((size_t)Nn * 4);
  float*          av    = (float*)carve((size_t)Nn * 4);
  unsigned int*   emax  = (unsigned int*)carve((size_t)Bb * 4);
  float*          denom = (float*)carve((size_t)Bb * 4);
  float*          rbuf  = (float*)carve((size_t)Bb * 128 * 4);
  float*          fpproj= (float*)carve((size_t)Bb * 128 * 4);
  unsigned short* comb  = (unsigned short*)carve((size_t)Bb * 384 * 2);
  unsigned short* t1b   = (unsigned short*)carve((size_t)Bb * 320 * 2);
  unsigned short* t2b   = (unsigned short*)carve((size_t)Bb * 256 * 2);
  (void)ws_size; (void)n_in; (void)out_size;

  // ---- weight transposes / input conversion ----
  auto cvtT = [&](const float* s, unsigned short* d, int Nreal, int Kreal,
                  int Npad, int Kpad, int swap64) {
    k_f2bt<<<nb((size_t)Npad * Kpad), 256, 0, stream>>>(s, d, Nreal, Kreal, Npad, Kpad, swap64);
  };
  cvtT(We2, We2b, 4096, 128, 4096, 128, /*swap64=*/1);   // -> per-edge [o,i] layout
  cvtT(Wres, Wresb, 64, 64, 64, 64, 0);
  cvtT(Wgx, Wgxb, 192, 64, 192, 64, 0);
  cvtT(Wgh, Wghb, 192, 64, 192, 64, 0);
  cvtT(Wlx[0], Wlxb[0], 512, 256, 512, 256, 0);
  cvtT(Wlh[0], Wlhb[0], 512, 128, 512, 128, 0);
  cvtT(Wlx[1], Wlxb[1], 512, 128, 512, 128, 0);
  cvtT(Wlh[1], Wlhb[1], 512, 128, 512, 128, 0);
  cvtT(Wlx[2], Wlxb[2], 512, 128, 512, 128, 0);
  cvtT(Wlh[2], Wlhb[2], 512, 128, 512, 128, 0);
  cvtT(Wfp, Wfpb, 128, 1024, 128, 1024, 0);
  cvtT(Wf1, Wf1b, 300, 384, 320, 384, 0);
  cvtT(Wf2, Wf2b, 256, 300, 256, 320, 0);
  cvtT(Wp, Wpb, 138, 256, 144, 256, 0);
  k_f2b<<<nb((size_t)Bb * 1024), 256, 0, stream>>>(fp_vector, fpb, (size_t)Bb * 1024);

  // ---- node projection + edge nets ----
  k_node_proj<<<nb((size_t)Nn * 64), 256, 0, stream>>>(node_feats, Wn, bn_, h, hb, Nn);
  k_edge_hidden<<<nb((size_t)E * 128), 256, 0, stream>>>(edge_feats, We1, be1, hiddenb, E);
  k_edge_emb<<<nb((size_t)E * 64), 256, 0, stream>>>(edge_feats, Wep, bep, eemb, E);

  // ---- big edge-network GEMM: Wt[E,4096] = hidden @ We2(perm) + be2 (bf16) ----
  // NOTE: be2 must be permuted the same way as We2 columns; be2 is zero in the
  // reference init but handle it properly via a permuted epilogue is avoided by
  // observing bias[n'] where n'=o*64+i must read be2[i*64+o]. We pre-permute it:
  // (cheap: done by reusing k_f2bt on a 1-row matrix is not possible; instead
  //  the permutation is folded by indexing bias with swap inside a tiny kernel.)
  // Here: copy permuted be2 into gbuf-front (4096 floats fit? gbuf=Bb*512=409600 ok
  // but gbuf is reused later; use a dedicated slice of ws): we reuse 'gx' buffer
  // front (Nn*192 >= 4096) before its first real use.
  {
    float* be2p = gx;  // scratch: permuted bias, consumed immediately by the GEMM
    // permute kernel inline via k_f2bt is bf16; need f32 -> small dedicated launch:
    // reuse k_zero32 pattern not possible; use a lambda kernel below.
    extern __global__ void k_permute_bias64(const float*, float*, int);
    k_permute_bias64<<<nb(4096), 256, 0, stream>>>(be2, be2p, 4096);
    gemm(stream, hiddenb, We2b, be2p, nullptr, Webf, E, 4096, 128,
         4096, 4096, 4096, /*relu=*/0, /*accum=*/0);
  }

  // ---- 3 GRU message-passing iterations ----
  for (int it = 0; it < 3; ++it) {
    k_zero32<<<nb((size_t)Nn * 64), 256, 0, stream>>>((unsigned int*)agg, (size_t)Nn * 64);
    k_msg_scatter<<<(E + 7) / 8, 256, 0, stream>>>(Webf, h, src, dst, agg, E);
    gemm(stream, hb, Wresb, nullptr, hres, nullptr, Nn, 64, 64, 64, 64, 64, 0, 0);
    k_gru_m<<<nb((size_t)Nn * 64), 256, 0, stream>>>(agg, hres, mb, (size_t)Nn * 64);
    gemm(stream, mb, Wgxb, bgx, gx, nullptr, Nn, 192, 64, 192, 192, 192, 0, 0);
    gemm(stream, hb, Wghb, bgh, gh, nullptr, Nn, 192, 64, 192, 192, 192, 0, 0);
    k_gru_update<<<nb((size_t)Nn * 64), 256, 0, stream>>>(gx, gh, h, hb, Nn);
  }

  // ---- readout: feat[N,128] = segment_sum(concat(h[src], e_emb), dst) ----
  k_zero32<<<nb((size_t)Nn * 128), 256, 0, stream>>>((unsigned int*)feat, (size_t)Nn * 128);
  k_readout<<<nb((size_t)E * 128), 256, 0, stream>>>(h, eemb, src, dst, feat, E);

  // ---- Set2Set state init (q_star = 0, h/c = 0) ----
  for (int l = 0; l < 3; ++l) {
    k_zero32<<<nb((size_t)Bb * 128), 256, 0, stream>>>((unsigned int*)hl[l], (size_t)Bb * 128);
    k_zero32<<<nb((size_t)Bb * 128), 256, 0, stream>>>((unsigned int*)cl[l], (size_t)Bb * 128);
    k_zero32<<<nb((size_t)Bb * 64), 256, 0, stream>>>((unsigned int*)hlb[l], (size_t)Bb * 64);
  }
  k_zero32<<<nb((size_t)Bb * 256), 256, 0, stream>>>((unsigned int*)qs, (size_t)Bb * 256);
  k_zero32<<<nb((size_t)Bb * 128), 256, 0, stream>>>((unsigned int*)qsb, (size_t)Bb * 128);

  // ---- 6 Set2Set iterations ----
  for (int it = 0; it < 6; ++it) {
    for (int l = 0; l < 3; ++l) {
      const unsigned short* xin = (l == 0) ? qsb : hlb[l - 1];
      int Kin = (l == 0) ? 256 : 128;
      gemm(stream, xin, Wlxb[l], bl[l], gbuf, nullptr, Bb, 512, Kin, 512, 512, 512, 0, 0);
      gemm(stream, hlb[l], Wlhb[l], nullptr, gbuf, nullptr, Bb, 512, 128, 512, 512, 512,
           0, /*accum=*/1);
      k_lstm_cell<<<nb((size_t)Bb * 128), 256, 0, stream>>>(gbuf, hl[l], cl[l], hlb[l], Bb);
    }
    k_init_negmax<<<nb(Bb), 256, 0, stream>>>(emax, Bb);
    k_zero32<<<nb(Bb), 256, 0, stream>>>((unsigned int*)denom, Bb);
    k_zero32<<<nb((size_t)Bb * 128), 256, 0, stream>>>((unsigned int*)rbuf, (size_t)Bb * 128);
    k_attn_dot<<<nb(Nn), 256, 0, stream>>>(feat, hl[2], gid, ev, emax, Nn);
    k_attn_exp<<<nb(Nn), 256, 0, stream>>>(ev, emax, gid, av, denom, Nn);
    k_attn_r<<<nb((size_t)Nn * 128), 256, 0, stream>>>(feat, av, denom, gid, rbuf, Nn);
    k_qstar<<<nb((size_t)Bb * 256), 256, 0, stream>>>(hl[2], rbuf, qs, qsb, Bb);
  }

  // ---- fingerprint encoder + combined FFN + head ----
  gemm(stream, fpb, Wfpb, bfp, fpproj, nullptr, Bb, 128, 1024, 128, 128, 128, 0, 0);
  k_build_comb<<<nb((size_t)Bb * 384), 256, 0, stream>>>(qs, fpproj, bn_g, bn_b, bn_m, bn_v,
                                                         comb, Bb);
  gemm(stream, comb, Wf1b, bf1, nullptr, t1b, Bb, 320, 384, 300, 320, 320, /*relu=*/1, 0);
  gemm(stream, t1b, Wf2b, bf2, nullptr, t2b, Bb, 256, 320, 256, 256, 256, /*relu=*/1, 0);
  gemm(stream, t2b, Wpb, bp, (float*)d_out, nullptr, Bb, 144, 256, 138, 138, 138, 0, 0);
}

// bias permute for the edge GEMM: out[o*64+i] = in[i*64+o]
__global__ void k_permute_bias64(const float* __restrict__ in, float* __restrict__ out, int n)
{
  int t = blockIdx.x * blockDim.x + threadIdx.x;
  if (t >= n) return;
  out[((t & 63) << 6) | (t >> 6)] = in[t];
}
